// VoCoHead_88974542504179
// MI455X (gfx1250) — matrix operand order repaired and checked by reading, not compiled, not run on hardware
//
#include <hip/hip_runtime.h>
#include <math.h>

typedef unsigned short u16;
typedef unsigned int   u32;

typedef __attribute__((ext_vector_type(16))) __bf16 v16bf;
typedef __attribute__((ext_vector_type(8)))  float  v8f;

struct Q2 { uint4 lo, hi; };   // 32 bytes == one 16-element bf16 fragment

// ---------------- problem constants ----------------
#define Bb    32
#define Ss    256
#define Kk    16
#define Nn    272            // S + K rows per batch
#define CIN   768
#define HH    2048
#define MROWS (Bb*Nn)        // 8704 = 68 * 128

__device__ __forceinline__ u16 f2bf(float f) {
  u32 u = __float_as_uint(f);
  u32 r = u + 0x7FFFu + ((u >> 16) & 1u);   // round-to-nearest-even
  return (u16)(r >> 16);
}

// async 16-byte global -> LDS copy (per-lane), ASYNCcnt-tracked (CDNA5)
__device__ __forceinline__ void async_ld16(u32 ldsOff, u32 gByteOff, const void* base) {
  asm volatile("global_load_async_to_lds_b128 %0, %1, %2"
               :: "v"(ldsOff), "v"(gByteOff), "s"(base)
               : "memory");
}
__device__ __forceinline__ void wait_async0() {
  asm volatile("s_wait_asynccnt 0x0" ::: "memory");
}

// ---------------- pack xb = concat(x, bases) per batch, fp32 -> bf16 --------
__global__ void pack_xb(const float* __restrict__ xf, const float* __restrict__ bsf,
                        u16* __restrict__ xb) {
  int idx = blockIdx.x * blockDim.x + threadIdx.x;   // one per 4 elements
  int e = idx * 4;
  if (e >= MROWS * CIN) return;
  int row = e / CIN, c = e % CIN;
  int b = row / Nn, n = row % Nn;
  const float* src = (n < Ss) ? (xf  + (size_t)(b * Ss + n)        * CIN + c)
                              : (bsf + (size_t)(b * Kk + (n - Ss)) * CIN + c);
  float4 v = *(const float4*)src;
  ushort4 o; o.x = f2bf(v.x); o.y = f2bf(v.y); o.z = f2bf(v.z); o.w = f2bf(v.w);
  *(ushort4*)(xb + e) = o;
}

// ---------------- weight transpose-convert: W[K][N] fp32 -> Wt[N][K] bf16 ---
__global__ void wconv_t(const float* __restrict__ W, u16* __restrict__ Wt,
                        int Kg, int Ng) {
  int idx = blockIdx.x * blockDim.x + threadIdx.x;
  if (idx >= Kg * Ng) return;
  int k = idx / Ng, n = idx % Ng;
  Wt[(size_t)n * Kg + k] = f2bf(W[idx]);
}

// ---------------- WMMA bf16 GEMM: Y[M][N] = A[M][K] * Bt[N][K]^T + bias -----
// block tile 128x128, 8 waves (4x2), wave tile 32x64 = 2x4 wmma 16x16 tiles.
// global->LDS staging uses GLOBAL_LOAD_ASYNC_TO_LDS_B128 double-buffered.
#define LDK 40   // LDS row stride in bf16 elements (32 data + 8 pad)

__global__ __launch_bounds__(256) void gemm_bf16(
    const u16* __restrict__ A, const u16* __restrict__ Bt,
    const float* __restrict__ bias, float* __restrict__ Y,
    int Ng, int Kg) {
  __shared__ u16 sA[2][128 * LDK];
  __shared__ u16 sB[2][128 * LDK];
  const int t    = threadIdx.x;
  const int lane = t & 31;
  const int wave = t >> 5;
  const int wm   = wave >> 1;          // 0..3
  const int wn   = wave & 1;           // 0..1
  const int m0   = blockIdx.y * 128;
  const int n0   = blockIdx.x * 128;

  // cooperative tile staging: 256 threads x 2 x 16B async copies per matrix
  const int lr = t >> 2;               // 0..63
  const int lq = t & 3;                // 0..3 -> k offset lq*8 elems
  const u32 rowBytes = (u32)Kg * 2;
  u32 gaOff0 = (u32)(m0 + lr) * rowBytes + (u32)(lq * 16);
  u32 gaOff1 = gaOff0 + 64u * rowBytes;
  u32 gbOff0 = (u32)(n0 + lr) * rowBytes + (u32)(lq * 16);
  u32 gbOff1 = gbOff0 + 64u * rowBytes;
  const int sOff0 = lr * LDK + lq * 8;
  const int sOff1 = sOff0 + 64 * LDK;

  u32 lA0[2], lA1[2], lB0[2], lB1[2];
  #pragma unroll
  for (int s = 0; s < 2; ++s) {
    lA0[s] = (u32)(size_t)&sA[s][sOff0];
    lA1[s] = (u32)(size_t)&sA[s][sOff1];
    lB0[s] = (u32)(size_t)&sB[s][sOff0];
    lB1[s] = (u32)(size_t)&sB[s][sOff1];
  }

  v8f zero;
  #pragma unroll
  for (int i = 0; i < 8; ++i) zero[i] = 0.0f;
  v8f acc[2][4];
  #pragma unroll
  for (int i = 0; i < 2; ++i)
    #pragma unroll
    for (int j = 0; j < 4; ++j) acc[i][j] = zero;

  const int steps = Kg >> 5;
  // prologue: stage 0 (async)
  async_ld16(lA0[0], gaOff0, A);
  async_ld16(lA1[0], gaOff1, A);
  async_ld16(lB0[0], gbOff0, Bt);
  async_ld16(lB1[0], gbOff1, Bt);

  const int h   = lane >> 4;    // half-wave select
  const int l16 = lane & 15;

  int cur = 0;
  for (int ks = 0; ks < steps; ++ks) {
    wait_async0();       // my wave's async writes into buf[cur] landed
    __syncthreads();     // everyone's landed
    const bool more = (ks + 1) < steps;
    if (more) {
      int nxt = cur ^ 1;
      u32 ko = (u32)(ks + 1) * 64u;    // 32 bf16 elems = 64 bytes per K-step
      async_ld16(lA0[nxt], gaOff0 + ko, A);
      async_ld16(lA1[nxt], gaOff1 + ko, A);
      async_ld16(lB0[nxt], gbOff0 + ko, Bt);
      async_ld16(lB1[nxt], gbOff1 + ko, Bt);
    }
    const u16* ab = sA[cur];
    const u16* bb = sB[cur];
    v16bf af[2], bfr[4];
    // A fragment: lane l16 holds row m; a[i] = A[m][K0 + i%8 + 8h + 16*(i/8)]
    #pragma unroll
    for (int mt = 0; mt < 2; ++mt) {
      const u16* p = ab + (wm * 32 + mt * 16 + l16) * LDK + 8 * h;
      Q2 q; q.lo = *(const uint4*)p; q.hi = *(const uint4*)(p + 16);
      af[mt] = __builtin_bit_cast(v16bf, q);
    }
    // B fragment: lane l16 holds col n; b[i] = B[K0 + i%8 + 8h + 16*(i/8)][n]
    #pragma unroll
    for (int nt = 0; nt < 4; ++nt) {
      const u16* p = bb + (wn * 64 + nt * 16 + l16) * LDK + 8 * h;
      Q2 q; q.lo = *(const uint4*)p; q.hi = *(const uint4*)(p + 16);
      bfr[nt] = __builtin_bit_cast(v16bf, q);
    }
    #pragma unroll
    for (int mt = 0; mt < 2; ++mt)
      #pragma unroll
      for (int nt = 0; nt < 4; ++nt)
        acc[mt][nt] = __builtin_amdgcn_wmma_f32_16x16x32_bf16(
            false, af[mt], false, bfr[nt], (short)0, acc[mt][nt], false, false);
    cur ^= 1;
  }

  // epilogue: C/D layout -> M = r + 8*(lane/16), N = lane%16
  #pragma unroll
  for (int mt = 0; mt < 2; ++mt) {
    #pragma unroll
    for (int nt = 0; nt < 4; ++nt) {
      int col = n0 + wn * 64 + nt * 16 + l16;
      float bv = bias[col];
      #pragma unroll
      for (int r = 0; r < 8; ++r) {
        int row = m0 + wm * 32 + mt * 16 + 8 * h + r;
        Y[(size_t)row * Ng + col] = acc[mt][nt][r] + bv;
      }
    }
  }
}

// ---------------- batchnorm over axis=1 (272 rows per batch) ----------------
__global__ void bn_stats(const float* __restrict__ Y, const float* __restrict__ g,
                         const float* __restrict__ be, float* __restrict__ scale,
                         float* __restrict__ shift) {
  int id = blockIdx.x * blockDim.x + threadIdx.x;
  if (id >= Bb * HH) return;
  int b = id >> 11, ch = id & (HH - 1);
  const float* p = Y + (size_t)(b * Nn) * HH + ch;
  float s = 0.f, s2 = 0.f;
  for (int n = 0; n < Nn; ++n) { float v = *p; s += v; s2 += v * v; p += HH; }
  float m   = s  * (1.0f / Nn);
  float var = s2 * (1.0f / Nn) - m * m;
  float sc  = g[ch] * rsqrtf(var + 1e-5f);
  scale[id] = sc;
  shift[id] = be[ch] - m * sc;
}

__global__ void bn_apply_relu(const float* __restrict__ Y, const float* __restrict__ scale,
                              const float* __restrict__ shift, u16* __restrict__ Hb) {
  int idx = blockIdx.x * blockDim.x + threadIdx.x;
  int e = idx * 4;
  if (e >= MROWS * HH) return;
  int row = e >> 11;
  int c   = e & (HH - 1);
  int b   = row / Nn;
  int sidx = (b << 11) + c;
  float4 v  = *(const float4*)(Y + e);
  float4 sc = *(const float4*)(scale + sidx);
  float4 sh = *(const float4*)(shift + sidx);
  ushort4 o;
  o.x = f2bf(fmaxf(v.x * sc.x + sh.x, 0.f));
  o.y = f2bf(fmaxf(v.y * sc.y + sh.y, 0.f));
  o.z = f2bf(fmaxf(v.z * sc.z + sh.z, 0.f));
  o.w = f2bf(fmaxf(v.w * sc.w + sh.w, 0.f));
  *(ushort4*)(Hb + e) = o;
}

// ---------------- L2-normalize each row (in place) --------------------------
__global__ __launch_bounds__(256) void l2norm_rows(float* __restrict__ Y) {
  __shared__ float sh[256];
  int row = blockIdx.x;
  int t = threadIdx.x;
  float* p = Y + (size_t)row * HH;
  float v[8]; float s = 0.f;
  #pragma unroll
  for (int i = 0; i < 8; ++i) { v[i] = p[t + i * 256]; s += v[i] * v[i]; }
  sh[t] = s; __syncthreads();
  for (int w = 128; w > 0; w >>= 1) { if (t < w) sh[t] += sh[t + w]; __syncthreads(); }
  float inv = 1.0f / fmaxf(sqrtf(sh[0]), 1e-8f);
  #pragma unroll
  for (int i = 0; i < 8; ++i) p[t + i * 256] = v[i] * inv;
}

// ---------------- logits + pos/neg loss accumulation ------------------------
__global__ __launch_bounds__(256) void logits_loss(
    const float* __restrict__ P, const float* __restrict__ labels,
    float* __restrict__ acc) {
  __shared__ float sh[256];
  __shared__ float tsh[64];
  int b = blockIdx.x >> 8;
  int s = blockIdx.x & 255;
  int t = threadIdx.x;
  int k   = t >> 4;
  int i16 = t & 15;
  const float* xr = P + (size_t)(b * Nn + s) * HH;
  const float* br = P + (size_t)(b * Nn + Ss + k) * HH;
  float part = 0.f;
  for (int j = i16; j < HH; j += 16) part += xr[j] * br[j];
  sh[t] = part; __syncthreads();
  if (t < 16) {
    float dot = 0.f;
    #pragma unroll
    for (int i = 0; i < 16; ++i) dot += sh[t * 16 + i];
    float logit = fmaxf(dot, 0.f);
    float lab = labels[(size_t)(b * Ss + s) * Kk + t];
    float pl = (lab > 0.f)  ? 1.f : 0.f;
    float nl = (lab == 0.f) ? 1.f : 0.f;
    float pdis = fabsf(lab - logit);
    tsh[t]      = pl * logf(1.f - pdis + 1e-6f);
    tsh[16 + t] = pl;
    tsh[32 + t] = nl * logit * logit;
    tsh[48 + t] = nl;
  }
  __syncthreads();
  if (t == 0) {
    float a0 = 0, a1 = 0, a2 = 0, a3 = 0;
    for (int i = 0; i < 16; ++i) {
      a0 += tsh[i]; a1 += tsh[16 + i]; a2 += tsh[32 + i]; a3 += tsh[48 + i];
    }
    atomicAdd(&acc[b],      a0);
    atomicAdd(&acc[32 + b], a1);
    atomicAdd(&acc[64 + b], a2);
    atomicAdd(&acc[96 + b], a3);
  }
}

// ---------------- sim = bn . bn^T, masked upper-tri reg ---------------------
__global__ __launch_bounds__(256) void sim_reg(const float* __restrict__ P,
                                               float* __restrict__ acc) {
  __shared__ float sh[256];
  int b = blockIdx.x, t = threadIdx.x;
  int i = t >> 4, j = t & 15;
  const float* bi = P + (size_t)(b * Nn + Ss + i) * HH;
  const float* bj = P + (size_t)(b * Nn + Ss + j) * HH;
  float dot = 0.f;
  for (int hh = 0; hh < HH; ++hh) dot += bi[hh] * bj[hh];
  float r = fmaxf(dot, 0.f);
  sh[t] = (j > i) ? r * r : 0.f;
  __syncthreads();
  for (int w = 128; w > 0; w >>= 1) { if (t < w) sh[t] += sh[t + w]; __syncthreads(); }
  if (t == 0) acc[128 + b] = sh[0] * (1.0f / 120.0f);   // K*(K-1)/2 = 120
}

__global__ void zero_acc(float* acc) {
  int t = threadIdx.x;
  if (t < 160) acc[t] = 0.f;
}

__global__ void finalize(const float* __restrict__ acc, float* __restrict__ out) {
  if (threadIdx.x == 0) {
    float p = 0.f, n = 0.f, r = 0.f;
    for (int b = 0; b < Bb; ++b) {
      p += -acc[b]        / (acc[32 + b] + 1e-6f);
      n +=  acc[64 + b]   / (acc[96 + b] + 1e-6f);
      r +=  acc[128 + b];
    }
    out[0] = p * (1.0f / Bb);
    out[1] = n * (1.0f / Bb);
    out[2] = r * (1.0f / Bb);
  }
}

// ---------------- orchestration ---------------------------------------------
extern "C" void kernel_launch(void* const* d_in, const int* in_sizes, int n_in,
                              void* d_out, int out_size, void* d_ws, size_t ws_size,
                              hipStream_t stream) {
  const float* xf   = (const float*)d_in[0];
  const float* bsf  = (const float*)d_in[1];
  const float* labs = (const float*)d_in[2];
  const float* W1   = (const float*)d_in[3];
  const float* b1   = (const float*)d_in[4];
  const float* g1v  = (const float*)d_in[5];
  const float* be1  = (const float*)d_in[6];
  const float* W2   = (const float*)d_in[7];
  const float* b2   = (const float*)d_in[8];
  const float* g2v  = (const float*)d_in[9];
  const float* be2  = (const float*)d_in[10];
  const float* W3   = (const float*)d_in[11];
  const float* b3   = (const float*)d_in[12];
  float* out = (float*)d_out;

  char* wsb = (char*)d_ws;
  size_t off = 0;
  auto take = [&](size_t bytes) -> char* {
    char* p = wsb + off;
    off += (bytes + 255) & ~(size_t)255;
    return p;
  };
  u16*   XB  = (u16*)  take((size_t)MROWS * CIN * 2);
  u16*   W1T = (u16*)  take((size_t)HH * CIN * 2);
  u16*   W2T = (u16*)  take((size_t)HH * HH * 2);
  u16*   W3T = (u16*)  take((size_t)HH * HH * 2);
  float* Y   = (float*)take((size_t)MROWS * HH * 4);
  u16*   HB  = (u16*)  take((size_t)MROWS * HH * 2);
  float* SC  = (float*)take((size_t)Bb * HH * 4);
  float* SH  = (float*)take((size_t)Bb * HH * 4);
  float* ACC = (float*)take(160 * 4);

  dim3 blk(256);
  pack_xb<<<(MROWS * CIN / 4 + 255) / 256, blk, 0, stream>>>(xf, bsf, XB);
  wconv_t<<<(CIN * HH + 255) / 256, blk, 0, stream>>>(W1, W1T, CIN, HH);
  wconv_t<<<(HH * HH + 255) / 256, blk, 0, stream>>>(W2, W2T, HH, HH);
  wconv_t<<<(HH * HH + 255) / 256, blk, 0, stream>>>(W3, W3T, HH, HH);
  zero_acc<<<1, blk, 0, stream>>>(ACC);

  dim3 gemmGrid(HH / 128, MROWS / 128);
  gemm_bf16<<<gemmGrid, blk, 0, stream>>>(XB, W1T, b1, Y, HH, CIN);
  bn_stats<<<(Bb * HH) / 256, blk, 0, stream>>>(Y, g1v, be1, SC, SH);
  bn_apply_relu<<<(MROWS * HH / 4) / 256, blk, 0, stream>>>(Y, SC, SH, HB);

  gemm_bf16<<<gemmGrid, blk, 0, stream>>>(HB, W2T, b2, Y, HH, HH);
  bn_stats<<<(Bb * HH) / 256, blk, 0, stream>>>(Y, g2v, be2, SC, SH);
  bn_apply_relu<<<(MROWS * HH / 4) / 256, blk, 0, stream>>>(Y, SC, SH, HB);

  gemm_bf16<<<gemmGrid, blk, 0, stream>>>(HB, W3T, b3, Y, HH, HH);

  l2norm_rows<<<MROWS, blk, 0, stream>>>(Y);
  logits_loss<<<Bb * Ss, blk, 0, stream>>>(Y, labs, ACC);
  sim_reg<<<Bb, blk, 0, stream>>>(Y, ACC);
  finalize<<<1, 32, 0, stream>>>(ACC, out);
}
